// GRU_Resnet_kan_70712341561500
// MI455X (gfx1250) — matrix operand order repaired and checked by reading, not compiled, not run on hardware
//
#include <hip/hip_runtime.h>
#include <math.h>
#include <string.h>

// ---------------------------------------------------------------------------
// Types for CDNA5 WMMA (wave32): v_wmma_f32_16x16x32_bf16
// ---------------------------------------------------------------------------
typedef __bf16 v16bf __attribute__((ext_vector_type(16)));
typedef float  v8f   __attribute__((ext_vector_type(8)));

#define HIDDEN 128
#define G3     384   // 3*HIDDEN
#define BATCH  512
#define SEQ    512
#define FEAT   32
#define TILE_B 16
#define NWAVES 12    // 24 N-tiles of 16 -> 2 per wave
#define NTHREADS (NWAVES * 32)
#define NTILES_B (BATCH / TILE_B)   // 32
#define LAYERS 4

// ---------------------------------------------------------------------------
// fp32 -> bf16 conversion (weights)
// ---------------------------------------------------------------------------
__global__ void f32_to_bf16_kernel(const float* __restrict__ in,
                                   __bf16* __restrict__ out, int n) {
  int i = blockIdx.x * blockDim.x + threadIdx.x;
  if (i < n) out[i] = (__bf16)in[i];
}

__global__ void zero_i32_kernel(int* __restrict__ p, int n) {
  int i = blockIdx.x * blockDim.x + threadIdx.x;
  if (i < n) p[i] = 0;
}

// ---------------------------------------------------------------------------
// One GRU layer body for a 16-row batch tile, iterating all timesteps.
// Per step: gi = x_t @ Wih^T + b_ih ; gh = h @ Whh^T + b_hh   (WMMA bf16)
// then f32 gate math in LDS, h update, bf16 sequence output.
// Cross-layer pipeline sync via acquire/release progress counters.
// ---------------------------------------------------------------------------
template <int FIN>
__device__ __forceinline__
void gru_layer_body(const float*  __restrict__ x_f32,     // FIN==32 input path
                    const __bf16* __restrict__ in_seq,    // FIN==128 input path
                    const __bf16* __restrict__ w_ih,      // (384,FIN) bf16
                    const __bf16* __restrict__ w_hh,      // (384,128) bf16
                    const float*  __restrict__ b_ih,      // (384)
                    const float*  __restrict__ b_hh,      // (384)
                    __bf16*       __restrict__ out_seq,   // (B,S,128) bf16
                    float*        __restrict__ last_out,  // (B,128) or null
                    int*          prog_in,                // upstream progress or null
                    int*          prog_out,               // this layer's progress
                    int bbase, int S,
                    __bf16* sh_h, __bf16* sh_x, float* s_gi, float* s_gh) {
  constexpr int KIN = FIN / 32;

  const int tid  = threadIdx.x;
  const int lane = tid & 31;
  const int wid  = tid >> 5;                 // 0..11

  // B-fragment (bf16 32x16): lane -> N = lane&15, K = (lane<16?0:16)+2v+p
  // A-fragment (bf16 16x32): lane -> M = lane&15,
  //   K = (lane<16?0:16)+2v+p
  const int n_loc = lane & 15;
  const int bkb   = (lane < 16) ? 0 : 16;
  const int akb   = (lane < 16) ? 0 : 8;
  const int m_loc = lane & 15;

  // Preload weight B-fragments (constant over the whole sequence)
  v16bf Bih[KIN][2];
  v16bf Bhh[4][2];
  float bias_i[2], bias_h[2];
#pragma unroll
  for (int nt2 = 0; nt2 < 2; ++nt2) {
    const int nt  = 2 * wid + nt2;
    const int row = nt * 16 + n_loc;
    bias_i[nt2] = b_ih[row];
    bias_h[nt2] = b_hh[row];
#pragma unroll
    for (int kt = 0; kt < KIN; ++kt)
#pragma unroll
      for (int v = 0; v < 8; ++v)
#pragma unroll
        for (int p = 0; p < 2; ++p) {
          int k = kt * 32 + bkb + 2 * v + p;
          Bih[kt][nt2][2 * v + p] = w_ih[row * FIN + k];
        }
#pragma unroll
    for (int kt = 0; kt < 4; ++kt)
#pragma unroll
      for (int v = 0; v < 8; ++v)
#pragma unroll
        for (int p = 0; p < 2; ++p) {
          int k = kt * 32 + bkb + 2 * v + p;
          Bhh[kt][nt2][2 * v + p] = w_hh[row * HIDDEN + k];
        }
  }

  // h0 = 0
  for (int i = tid; i < TILE_B * HIDDEN; i += NTHREADS) sh_h[i] = (__bf16)0.0f;
  __syncthreads();

  for (int t = 0; t < S; ++t) {
    // ---- wait for upstream layer to publish step t ----
    if (prog_in) {
      if (tid == 0) {
        while (__hip_atomic_load(prog_in, __ATOMIC_ACQUIRE,
                                 __HIP_MEMORY_SCOPE_AGENT) <= t)
          __builtin_amdgcn_s_sleep(1);
      }
      __syncthreads();
    }

    // ---- stage input tile into LDS (bf16), prefetch next step ----
    if constexpr (FIN == 32) {
      for (int i = tid; i < TILE_B * FIN; i += NTHREADS) {
        int b = i / FIN, f = i % FIN;
        size_t base = ((size_t)(bbase + b) * S + t) * FIN + f;
        sh_x[i] = (__bf16)x_f32[base];
        if (t + 1 < S) __builtin_prefetch(&x_f32[base + FIN], 0, 1);
      }
    } else {
      // 16 rows x 128 bf16 = 256 x uint4
      for (int i = tid; i < TILE_B * FIN / 8; i += NTHREADS) {
        int b = i >> 4, c = i & 15;
        size_t elem = ((size_t)(bbase + b) * S + t) * FIN + c * 8;
        uint4 v4 = *(const uint4*)(in_seq + elem);
        ((uint4*)sh_x)[i] = v4;
        if (t + 1 < S) __builtin_prefetch(in_seq + elem + FIN, 0, 1);
      }
    }
    __syncthreads();

    // ---- WMMA GEMMs: acc_i = x_t @ Wih^T + b_ih ; acc_h = h @ Whh^T + b_hh
    v8f ai[2], ah[2];
#pragma unroll
    for (int nt2 = 0; nt2 < 2; ++nt2)
#pragma unroll
      for (int r = 0; r < 8; ++r) {
        ai[nt2][r] = bias_i[nt2];
        ah[nt2][r] = bias_h[nt2];
      }

#pragma unroll
    for (int kt = 0; kt < KIN; ++kt) {
      v16bf A;
#pragma unroll
      for (int v = 0; v < 8; ++v)
#pragma unroll
        for (int p = 0; p < 2; ++p) {
          int k = kt * 32 + bkb + 2 * v + p;
          A[2 * v + p] = sh_x[m_loc * FIN + k];
        }
#pragma unroll
      for (int nt2 = 0; nt2 < 2; ++nt2)
        ai[nt2] = __builtin_amdgcn_wmma_f32_16x16x32_bf16(
            false, A, false, Bih[kt][nt2], (short)0, ai[nt2], false, false);
    }

#pragma unroll
    for (int kt = 0; kt < 4; ++kt) {
      v16bf A;
#pragma unroll
      for (int v = 0; v < 8; ++v)
#pragma unroll
        for (int p = 0; p < 2; ++p) {
          int k = kt * 32 + bkb + 2 * v + p;
          A[2 * v + p] = sh_h[m_loc * HIDDEN + k];
        }
#pragma unroll
      for (int nt2 = 0; nt2 < 2; ++nt2)
        ah[nt2] = __builtin_amdgcn_wmma_f32_16x16x32_bf16(
            false, A, false, Bhh[kt][nt2], (short)0, ah[nt2], false, false);
    }

    // ---- spill accumulators (C layout: M = r + (lane<16?0:8), N = lane&15)
#pragma unroll
    for (int nt2 = 0; nt2 < 2; ++nt2) {
      const int nt  = 2 * wid + nt2;
      const int col = nt * 16 + n_loc;
#pragma unroll
      for (int r = 0; r < 8; ++r) {
        int row = r + ((lane < 16) ? 0 : 8);
        s_gi[row * G3 + col] = ai[nt2][r];
        s_gh[row * G3 + col] = ah[nt2][r];
      }
    }
    __syncthreads();

    // ---- gate math + h update (f32 VALU) ----
    for (int idx = tid; idx < TILE_B * HIDDEN; idx += NTHREADS) {
      int b = idx >> 7, j = idx & 127;
      float ir  = s_gi[b * G3 + j];
      float iz  = s_gi[b * G3 + HIDDEN + j];
      float inn = s_gi[b * G3 + 2 * HIDDEN + j];
      float hr  = s_gh[b * G3 + j];
      float hz  = s_gh[b * G3 + HIDDEN + j];
      float hn  = s_gh[b * G3 + 2 * HIDDEN + j];
      float r = 1.0f / (1.0f + expf(-(ir + hr)));
      float z = 1.0f / (1.0f + expf(-(iz + hz)));
      float n = tanhf(inn + r * hn);
      float hprev = (float)sh_h[idx];
      float hnew  = (1.0f - z) * n + z * hprev;
      sh_h[idx] = (__bf16)hnew;
      out_seq[((size_t)(bbase + b) * S + t) * HIDDEN + j] = (__bf16)hnew;
      if (last_out && t == S - 1)
        last_out[(size_t)(bbase + b) * HIDDEN + j] = hnew;
    }
    __syncthreads();

    // ---- publish step t to downstream layer ----
    if (tid == 0) {
      __threadfence();
      __hip_atomic_store(prog_out, t + 1, __ATOMIC_RELEASE,
                         __HIP_MEMORY_SCOPE_AGENT);
    }
  }
}

// ---------------------------------------------------------------------------
// Fused 4-layer pipelined GRU: 128 workgroups = 4 layers x 32 batch tiles.
// Layer l, tile bt runs step t as soon as layer l-1, tile bt published t.
// ---------------------------------------------------------------------------
__global__ __launch_bounds__(NTHREADS)
void gru_fused_kernel(const float*  __restrict__ x,        // (B,S,32)
                      const __bf16* __restrict__ wih0_bf,  // (384,32)
                      const __bf16* __restrict__ wih_bf,   // (3,384,128)
                      const __bf16* __restrict__ whh_bf,   // (4,384,128)
                      const float*  __restrict__ b_ih,     // (4,384)
                      const float*  __restrict__ b_hh,     // (4,384)
                      __bf16*       __restrict__ seqA,
                      __bf16*       __restrict__ seqB,
                      float*        __restrict__ hlast,    // (B,128)
                      int*          __restrict__ progress, // (4,32)
                      int S) {
  __shared__ __align__(16) __bf16 sh_h[TILE_B * HIDDEN];
  __shared__ __align__(16) __bf16 sh_x[TILE_B * HIDDEN];
  __shared__ float s_gi[TILE_B * G3];
  __shared__ float s_gh[TILE_B * G3];

  const int layer = blockIdx.x >> 5;     // 0..3
  const int bt    = blockIdx.x & 31;     // 0..31
  const int bbase = bt * TILE_B;

  int* prog_out = progress + layer * NTILES_B + bt;
  int* prog_in  = (layer > 0) ? (progress + (layer - 1) * NTILES_B + bt) : nullptr;

  if (layer == 0) {
    gru_layer_body<32>(x, nullptr, wih0_bf, whh_bf, b_ih, b_hh,
                       seqA, nullptr, prog_in, prog_out, bbase, S,
                       sh_h, sh_x, s_gi, s_gh);
  } else {
    // ping-pong: out(l) = even(l) ? A : B ; in(l) = out(l-1)
    const __bf16* in  = (layer & 1) ? seqA : seqB;
    __bf16*       out = (layer & 1) ? seqB : seqA;
    gru_layer_body<128>(nullptr, in,
                        wih_bf + (size_t)(layer - 1) * G3 * HIDDEN,
                        whh_bf + (size_t)layer * G3 * HIDDEN,
                        b_ih + layer * G3, b_hh + layer * G3,
                        out, (layer == 3) ? hlast : nullptr,
                        prog_in, prog_out, bbase, S,
                        sh_h, sh_x, s_gi, s_gh);
  }
}

// ---------------------------------------------------------------------------
// BatchNorm over batch dim of h[:, -1, :]   (512 x 128)
// ---------------------------------------------------------------------------
__global__ void bn_kernel(const float* __restrict__ h,
                          const float* __restrict__ gamma,
                          const float* __restrict__ beta,
                          float* __restrict__ out) {
  int j = threadIdx.x;  // 0..127
  float s = 0.0f, s2 = 0.0f;
  for (int b = 0; b < BATCH; ++b) {
    float v = h[b * HIDDEN + j];
    s += v; s2 += v * v;
  }
  float mu  = s * (1.0f / BATCH);
  float var = s2 * (1.0f / BATCH) - mu * mu;
  float inv = rsqrtf(var + 1e-5f);
  float g = gamma[j], be = beta[j];
  for (int b = 0; b < BATCH; ++b)
    out[b * HIDDEN + j] = (h[b * HIDDEN + j] - mu) * inv * g + be;
}

// ---------------------------------------------------------------------------
// KAN spline basis: uniform grid g[t] = (t-3)*0.4 - 1, order K=3, 8 bases
// ---------------------------------------------------------------------------
__device__ __forceinline__ void kan_basis(float x, float* v) {
  float g[12];
#pragma unroll
  for (int t = 0; t < 12; ++t) g[t] = (float)(t - 3) * 0.4f - 1.0f;
  float w[11];
#pragma unroll
  for (int t = 0; t < 11; ++t)
    w[t] = (x >= g[t] && x < g[t + 1]) ? 1.0f : 0.0f;
#pragma unroll
  for (int k = 1; k <= 3; ++k) {
#pragma unroll
    for (int t = 0; t < 11; ++t) {
      if (t < 11 - k)
        w[t] = (x - g[t]) / (g[t + k] - g[t]) * w[t] +
               (g[t + k + 1] - x) / (g[t + k + 1] - g[t + 1]) * w[t + 1];
    }
  }
#pragma unroll
  for (int k = 0; k < 8; ++k) v[k] = w[k];
}

__device__ __forceinline__ float silu_f(float x) { return x / (1.0f + expf(-x)); }

// KAN layer 0: (512,128) -> (512,64)
__global__ void kan0_kernel(const float* __restrict__ in,
                            const float* __restrict__ coef,    // (128,64,8)
                            const float* __restrict__ sb,      // (128,64)
                            const float* __restrict__ ssp,     // (128,64)
                            float* __restrict__ out) {         // (512,64)
  __shared__ float sbase[128];
  __shared__ float sbas[128][8];
  int b = blockIdx.x, o = threadIdx.x;  // 64 threads
  for (int i = o; i < 128; i += 64) {
    float x = in[b * 128 + i];
    sbase[i] = silu_f(x);
    float v[8];
    kan_basis(x, v);
#pragma unroll
    for (int k = 0; k < 8; ++k) sbas[i][k] = v[k];
  }
  __syncthreads();
  float y = 0.0f;
  for (int i = 0; i < 128; ++i) {
    float spl = 0.0f;
#pragma unroll
    for (int k = 0; k < 8; ++k) spl += sbas[i][k] * coef[(i * 64 + o) * 8 + k];
    y += sb[i * 64 + o] * sbase[i] + ssp[i * 64 + o] * spl;
  }
  out[b * 64 + o] = y;
}

// KAN layer 1: (512,64) -> (512,1) + sigmoid
__global__ void kan1_kernel(const float* __restrict__ in,
                            const float* __restrict__ coef,   // (64,1,8)
                            const float* __restrict__ sb,
                            const float* __restrict__ ssp,
                            float* __restrict__ out) {        // (512,)
  int b = threadIdx.x;  // 512 threads
  float y = 0.0f;
  for (int i = 0; i < 64; ++i) {
    float x = in[b * 64 + i];
    float v[8];
    kan_basis(x, v);
    float spl = 0.0f;
#pragma unroll
    for (int k = 0; k < 8; ++k) spl += v[k] * coef[i * 8 + k];
    y += sb[i] * silu_f(x) + ssp[i] * spl;
  }
  out[b] = 1.0f / (1.0f + expf(-y));
}

// ---------------------------------------------------------------------------
// Host launcher
// ---------------------------------------------------------------------------
extern "C" void kernel_launch(void* const* d_in, const int* in_sizes, int n_in,
                              void* d_out, int out_size, void* d_ws, size_t ws_size,
                              hipStream_t stream) {
  (void)in_sizes; (void)n_in; (void)out_size; (void)ws_size;
  const float* x        = (const float*)d_in[0];   // (512,512,32)
  const float* w_ih0    = (const float*)d_in[1];   // (384,32)
  const float* w_ih     = (const float*)d_in[2];   // (3,384,128)
  const float* w_hh     = (const float*)d_in[3];   // (4,384,128)
  const float* b_ih     = (const float*)d_in[4];   // (4,384)
  const float* b_hh     = (const float*)d_in[5];   // (4,384)
  const float* bn_gamma = (const float*)d_in[6];
  const float* bn_beta  = (const float*)d_in[7];
  const float* k0_coef  = (const float*)d_in[8];
  const float* k0_sb    = (const float*)d_in[9];
  const float* k0_ssp   = (const float*)d_in[10];
  const float* k1_coef  = (const float*)d_in[11];
  const float* k1_sb    = (const float*)d_in[12];
  const float* k1_ssp   = (const float*)d_in[13];

  // ---- workspace layout ----
  char* ws = (char*)d_ws;
  size_t off = 0;
  const size_t seq_bytes = (size_t)BATCH * SEQ * HIDDEN * sizeof(__bf16);  // 64MB
  __bf16* seqA = (__bf16*)(ws + off); off += seq_bytes;
  __bf16* seqB = (__bf16*)(ws + off); off += seq_bytes;
  __bf16* wih0_bf = (__bf16*)(ws + off); off += (size_t)G3 * FEAT * 2;
  __bf16* wih_bf  = (__bf16*)(ws + off); off += (size_t)3 * G3 * HIDDEN * 2;
  __bf16* whh_bf  = (__bf16*)(ws + off); off += (size_t)4 * G3 * HIDDEN * 2;
  float* hlast = (float*)(ws + off); off += (size_t)BATCH * HIDDEN * 4;
  float* bnout = (float*)(ws + off); off += (size_t)BATCH * HIDDEN * 4;
  float* k0out = (float*)(ws + off); off += (size_t)BATCH * 64 * 4;
  int* progress = (int*)(ws + off);  off += (size_t)LAYERS * NTILES_B * 4;

  // ---- weight conversion fp32 -> bf16, progress reset ----
  {
    int n0 = G3 * FEAT;
    int n1 = 3 * G3 * HIDDEN;
    int n2 = 4 * G3 * HIDDEN;
    f32_to_bf16_kernel<<<(n0 + 255) / 256, 256, 0, stream>>>(w_ih0, wih0_bf, n0);
    f32_to_bf16_kernel<<<(n1 + 255) / 256, 256, 0, stream>>>(w_ih, wih_bf, n1);
    f32_to_bf16_kernel<<<(n2 + 255) / 256, 256, 0, stream>>>(w_hh, whh_bf, n2);
    zero_i32_kernel<<<1, 128, 0, stream>>>(progress, LAYERS * NTILES_B);
  }

  // ---- fused pipelined GRU: 4 layers x 32 batch tiles = 128 workgroups ----
  gru_fused_kernel<<<LAYERS * NTILES_B, NTHREADS, 0, stream>>>(
      x, wih0_bf, wih_bf, whh_bf, b_ih, b_hh, seqA, seqB, hlast, progress, SEQ);

  // ---- BatchNorm over batch ----
  bn_kernel<<<1, HIDDEN, 0, stream>>>(hlast, bn_gamma, bn_beta, bnout);

  // ---- KAN layers ----
  kan0_kernel<<<BATCH, 64, 0, stream>>>(bnout, k0_coef, k0_sb, k0_ssp, k0out);
  kan1_kernel<<<1, BATCH, 0, stream>>>(k0out, k1_coef, k1_sb, k1_ssp, (float*)d_out);
}
